// recommendation_model_44882408243724
// MI455X (gfx1250) — matrix operand order repaired and checked by reading, not compiled, not run on hardware
//
#include <hip/hip_runtime.h>
#include <hip/hip_bf16.h>
#include <math.h>

typedef __attribute__((ext_vector_type(16))) _Float16 v16h;
typedef __attribute__((ext_vector_type(8)))  _Float16 v8h;
typedef __attribute__((ext_vector_type(8)))  float    v8f;
typedef __attribute__((ext_vector_type(4)))  unsigned int u32x4;
typedef __attribute__((ext_vector_type(4)))  int      i32x4;
typedef __attribute__((ext_vector_type(8)))  int      i32x8;

#define HALF_DIM 32
#define INT_EMBED 64
#define DMODEL 128     // concat dim = 64 + 64
#define HIDDEN 128
#define SEQ 100
#define SEQ_PAD 112    // 7 tiles of 16
#define XS 136         // padded LDS row stride in halfs (272B = 68 dwords -> no bank conflicts)
#define BETA_C 0.7f

#if defined(__has_builtin)
#if __has_builtin(__builtin_amdgcn_tensor_load_to_lds) && __has_builtin(__builtin_amdgcn_s_wait_tensorcnt)
#define HAVE_TDM 1
#endif
#endif

// ---------------------------------------------------------------------------
// Stage A: per-item neighbor self-attention (reshape-not-transpose semantics,
// valid because M == half == 32). One wave32 per item, lane = feature.
// ---------------------------------------------------------------------------
__global__ void __launch_bounds__(256) poi_attn_kernel(
    const float* __restrict__ E,
    const int*   __restrict__ near_pois,
    float*       __restrict__ out,
    int item_num)
{
    int wave = blockIdx.x * (blockDim.x >> 5) + (threadIdx.x >> 5);
    int lane = threadIdx.x & 31;
    if (wave >= item_num) return;

    const int* nb = near_pois + (size_t)wave * 32;  // wave-uniform -> scalar loads
    float q = E[(size_t)wave * HALF_DIM + lane];

    float neigh[32];  // neigh[h][feature=lane]
#pragma unroll
    for (int h = 0; h < 32; ++h) {
        int idx = nb[h];
        neigh[h] = E[(size_t)idx * HALF_DIM + lane];  // coalesced per h
    }

    float score = 0.f;
#pragma unroll
    for (int h = 0; h < 32; ++h)
        score += __shfl(q, h, 32) * neigh[h];
    score *= 0.17677669529663687f;  // 1/sqrt(32)

    float mx = score;
#pragma unroll
    for (int off = 16; off > 0; off >>= 1)
        mx = fmaxf(mx, __shfl_xor(mx, off, 32));
    float e = __expf(score - mx);
    float sum = e;
#pragma unroll
    for (int off = 16; off > 0; off >>= 1)
        sum += __shfl_xor(sum, off, 32);
    float w = e / sum;

    float o = 0.f;
#pragma unroll
    for (int m = 0; m < 32; ++m)
        o += __shfl(w, m, 32) * neigh[m];

    out[(size_t)wave * HALF_DIM + lane] = o;
}

// ---------------------------------------------------------------------------
// Column-vector GEMM helper: out[0:112] = A(112 x 128 f16 LDS, row stride XS)
// @ bcol(128x16, only column 0 meaningful). One wave; lanes with N==0 hold
// the result column. 28 v_wmma per call.
// ---------------------------------------------------------------------------
__device__ __forceinline__ void colvec_gemm(const _Float16* Abase,
                                            const v16h bcol[4],
                                            float* outArr, int lane)
{
    const int hi = lane >> 4;
    for (int mt = 0; mt < SEQ_PAD / 16; ++mt) {
        const int row = mt * 16 + (lane & 15);
        v16h a[4];
#pragma unroll
        for (int kb = 0; kb < 4; ++kb) {
            const v8h* pa0 = (const v8h*)&Abase[row * XS + kb * 32 + hi * 16];
            const v8h* pa1 = (const v8h*)&Abase[row * XS + kb * 32 + hi * 16 + 8];
            v8h lo = *pa0, h8 = *pa1;
#pragma unroll
            for (int i = 0; i < 8; ++i) { a[kb][i] = lo[i]; a[kb][i + 8] = h8[i]; }
        }
        v8f c = {0.f, 0.f, 0.f, 0.f, 0.f, 0.f, 0.f, 0.f};
#pragma unroll
        for (int kb = 0; kb < 4; ++kb)
            c = __builtin_amdgcn_wmma_f32_16x16x32_f16(
                    false, a[kb], false, bcol[kb], (short)0, c, false, false);
        if ((lane & 15) == 0) {
#pragma unroll
            for (int r = 0; r < 8; ++r)
                outArr[mt * 16 + 8 * hi + r] = c[r];
        }
    }
}

// ---------------------------------------------------------------------------
// Stage B: per-batch fused MLP + masked power-softmax + prediction.
// One block (8 wave32) per batch element.
//  - W1 (f32, 64KB) staged into LDS by the Tensor Data Mover, overlapped with
//    the history gathers; B fragments converted f32->f16 in registers.
//  - phase 3: H = relu(X@W1 + b1)                  (8 waves, wmma)
//  - phase 4: logits = H@W2 (wave 0), s = X@ones (wave 1)   (wmma)
// ---------------------------------------------------------------------------
__global__ void __launch_bounds__(256) fused_mlp_kernel(
    const int*   __restrict__ history,
    const int*   __restrict__ target,
    const float* __restrict__ a_sus,
    const float* __restrict__ a_inf,
    const float* __restrict__ E_hist,
    const float* __restrict__ E_tgt,
    const float* __restrict__ W1,
    const float* __restrict__ b1,
    const float* __restrict__ W2,
    float*       __restrict__ out)
{
    __shared__ __align__(16) float    sW1f32[DMODEL * HIDDEN];  // raw W1, row-major
    __shared__ __align__(16) _Float16 sX[SEQ_PAD * XS];         // hist * tgt
    __shared__ __align__(16) _Float16 sH[SEQ_PAD * XS];         // relu(X@W1+b1)
    __shared__ _Float16 sW2h[HIDDEN];
    __shared__ float sTgt[DMODEL];
    __shared__ float sB1[HIDDEN];
    __shared__ float sS[SEQ_PAD];
    __shared__ float sLogit[SEQ_PAD];
    __shared__ float sExp[SEQ];
    __shared__ int   sHist[SEQ];

    const int b    = blockIdx.x;
    const int tid  = threadIdx.x;
    const int lane = tid & 31;
    const int wv   = tid >> 5;
    const int t    = target[b];

    // ---- phase 0: kick off TDM DMA of W1 (64KB f32) into LDS (wave 0) ----
#ifdef HAVE_TDM
    if (wv == 0) {
        unsigned long long ga = (unsigned long long)(const void*)W1;
        unsigned int lds_off =
            (unsigned int)__builtin_amdgcn_readfirstlane((int)(unsigned long long)(const void*)&sW1f32[0]);
        u32x4 g0;
        g0[0] = 1u;                                   // count=1, user mode, no gather
        g0[1] = lds_off;                              // lds_addr
        g0[2] = (unsigned int)(ga & 0xffffffffu);     // global_addr[31:0]
        g0[3] = (unsigned int)((ga >> 32) & 0x01ffffffu) | 0x80000000u; // ga[56:32] | type=2
        i32x8 g1;
        g1[0] = (int)(2u << 16);     // data_size=4B, workgroup_mask=0, no extras
        g1[1] = (int)(128u << 16);   // tensor_dim0 = 128 (low 16 in high half)
        g1[2] = (int)(128u << 16);   // tensor_dim0 hi=0 | tensor_dim1 = 128
        g1[3] = (int)(128u << 16);   // tensor_dim1 hi=0 | tile_dim0 = 128
        g1[4] = 128;                 // tile_dim1 = 128, tile_dim2 = 0
        g1[5] = 128;                 // tensor_dim0_stride = 128
        g1[6] = 0;
        g1[7] = 0;
        i32x4 z4 = {0, 0, 0, 0};
#if __clang_major__ >= 23
        i32x8 z8 = {0, 0, 0, 0, 0, 0, 0, 0};
        __builtin_amdgcn_tensor_load_to_lds(g0, g1, z4, z4, z8, 0);
#else
        __builtin_amdgcn_tensor_load_to_lds(g0, g1, z4, z4, 0);
#endif
    }
#else
    // fallback: coalesced vector copy of W1 into LDS
    for (int i = tid * 4; i < DMODEL * HIDDEN; i += 256 * 4)
        *(float4*)&sW1f32[i] = *(const float4*)&W1[i];
#endif

    // ---- phase 1: target vector, b1, W2 ----
    if (tid < DMODEL) {
        int d = tid;
        float v = (d < 64) ? E_tgt[(size_t)t * 64 + d]
                : (d < 96) ? a_inf[(size_t)t * 32 + (d - 64)]
                           : a_sus[(size_t)t * 32 + (d - 96)];
        sTgt[d]  = v;
        sB1[d]   = b1[d];
        sW2h[d]  = (_Float16)W2[d];
    }
    __syncthreads();   // sTgt ready

    // ---- phase 2: build X = hist * tgt (f16), overlapped with the TDM DMA ----
    for (int l = wv; l < SEQ_PAD; l += 8) {
        if (l < SEQ) {
            int h = history[(size_t)b * SEQ + l];
            if (lane == 0) sHist[l] = h;
            float x0 = E_hist[(size_t)h * 64 + lane]       * sTgt[lane];
            float x1 = E_hist[(size_t)h * 64 + 32 + lane]  * sTgt[32 + lane];
            float x2 = a_sus[(size_t)h * 32 + lane]        * sTgt[64 + lane];
            float x3 = a_inf[(size_t)h * 32 + lane]        * sTgt[96 + lane];
            sX[l * XS + lane]      = (_Float16)x0;
            sX[l * XS + 32 + lane] = (_Float16)x1;
            sX[l * XS + 64 + lane] = (_Float16)x2;
            sX[l * XS + 96 + lane] = (_Float16)x3;
        } else {
            sX[l * XS + lane]      = (_Float16)0.f;
            sX[l * XS + 32 + lane] = (_Float16)0.f;
            sX[l * XS + 64 + lane] = (_Float16)0.f;
            sX[l * XS + 96 + lane] = (_Float16)0.f;
        }
    }

#ifdef HAVE_TDM
    if (wv == 0) __builtin_amdgcn_s_wait_tensorcnt(0);  // W1 DMA complete
#endif
    __syncthreads();   // sX + sW1f32 visible to everyone

    // ---- phase 3: H = relu(X@W1 + b1); wave wv owns N-strip [wv*16, +16) ----
    const int ncol = wv * 16 + (lane & 15);
    const int hi   = lane >> 4;

    // B fragments straight from the f32 LDS image of W1 (convert in registers).
    // Lane holds K = hi*16 .. hi*16+15 of column ncol.
    v16h bfrag[4];
#pragma unroll
    for (int kb = 0; kb < 4; ++kb) {
#pragma unroll
        for (int i = 0; i < 16; ++i) {
            int k = kb * 32 + hi * 16 + i;
            bfrag[kb][i] = (_Float16)sW1f32[k * HIDDEN + ncol];
        }
    }

    const float b1n = sB1[ncol];

    for (int mt = 0; mt < SEQ_PAD / 16; ++mt) {
        const int row = mt * 16 + (lane & 15);
        v16h a[4];
#pragma unroll
        for (int kb = 0; kb < 4; ++kb) {
            const v8h* pa0 = (const v8h*)&sX[row * XS + kb * 32 + hi * 16];
            const v8h* pa1 = (const v8h*)&sX[row * XS + kb * 32 + hi * 16 + 8];
            v8h lo = *pa0, h8 = *pa1;
#pragma unroll
            for (int i = 0; i < 8; ++i) { a[kb][i] = lo[i]; a[kb][i + 8] = h8[i]; }
        }
        v8f c = {0.f, 0.f, 0.f, 0.f, 0.f, 0.f, 0.f, 0.f};
#pragma unroll
        for (int kb = 0; kb < 4; ++kb)
            c = __builtin_amdgcn_wmma_f32_16x16x32_f16(
                    false, a[kb], false, bfrag[kb], (short)0, c, false, false);
        // fused bias + relu, store strip of H (f16) for the second GEMM pass
#pragma unroll
        for (int r = 0; r < 8; ++r) {
            float hval = fmaxf(c[r] + b1n, 0.f);
            sH[(mt * 16 + 8 * hi + r) * XS + ncol] = (_Float16)hval;
        }
    }
    __syncthreads();

    // ---- phase 4: logits = H@W2 (wave 0); s = X@ones (wave 1) ----
    if (wv == 0) {
        const bool c0 = (lane & 15) == 0;
        v16h bcol[4];
#pragma unroll
        for (int kb = 0; kb < 4; ++kb)
#pragma unroll
            for (int i = 0; i < 16; ++i)
                bcol[kb][i] = c0 ? sW2h[kb * 32 + hi * 16 + i] : (_Float16)0.f;
        colvec_gemm(sH, bcol, sLogit, lane);
    } else if (wv == 1) {
        const bool c0 = (lane & 15) == 0;
        v16h bones[4];
#pragma unroll
        for (int kb = 0; kb < 4; ++kb)
#pragma unroll
            for (int i = 0; i < 16; ++i)
                bones[kb][i] = c0 ? (_Float16)1.f : (_Float16)0.f;
        colvec_gemm(sX, bones, sS, lane);
    }
    __syncthreads();

    // ---- phase 5: masked exp, pow(sum, beta), prediction, sigmoid ----
    if (wv == 0) {
        float esum = 0.f;
        for (int l = lane; l < SEQ; l += 32) {
            float ev = (sHist[l] != t) ? __expf(sLogit[l]) : 0.f;
            sExp[l] = ev;
            esum += ev;
        }
#pragma unroll
        for (int off = 16; off > 0; off >>= 1)
            esum += __shfl_xor(esum, off, 32);
        float denom = __powf(esum, BETA_C);
        float pred = 0.f;
        for (int l = lane; l < SEQ; l += 32)
            pred += sExp[l] * sS[l];   // same lane wrote sExp[l] above
#pragma unroll
        for (int off = 16; off > 0; off >>= 1)
            pred += __shfl_xor(pred, off, 32);
        pred /= denom;
        if (lane == 0) out[b] = 1.f / (1.f + __expf(-pred));
    }
}

// ---------------------------------------------------------------------------
extern "C" void kernel_launch(void* const* d_in, const int* in_sizes, int n_in,
                              void* d_out, int out_size, void* d_ws, size_t ws_size,
                              hipStream_t stream) {
    const int*   history   = (const int*)  d_in[0];
    const int*   target    = (const int*)  d_in[1];
    const int*   near_pois = (const int*)  d_in[2];
    const float* E_in      = (const float*)d_in[3];
    const float* E_out     = (const float*)d_in[4];
    const float* E_hist    = (const float*)d_in[5];
    const float* E_tgt     = (const float*)d_in[6];
    const float* W1        = (const float*)d_in[7];
    const float* b1        = (const float*)d_in[8];
    const float* W2        = (const float*)d_in[9];
    float*       out       = (float*)d_out;

    const int B        = in_sizes[1];
    const int item_num = in_sizes[2] / 32;

    float* a_sus = (float*)d_ws;
    float* a_inf = a_sus + (size_t)item_num * HALF_DIM;

    int blocksA = (item_num + 7) / 8;
    poi_attn_kernel<<<blocksA, 256, 0, stream>>>(E_in,  near_pois, a_sus, item_num);
    poi_attn_kernel<<<blocksA, 256, 0, stream>>>(E_out, near_pois, a_inf, item_num);

    fused_mlp_kernel<<<B, 256, 0, stream>>>(history, target, a_sus, a_inf,
                                            E_hist, E_tgt, W1, b1, W2, out);
}
